// CausalSelfAttention_53343493816396
// MI455X (gfx1250) — matrix operand order repaired and checked by reading, not compiled, not run on hardware
//
#include <hip/hip_runtime.h>

typedef __bf16 bf16;
typedef __attribute__((ext_vector_type(16))) __bf16 v16bf;
typedef __attribute__((ext_vector_type(8)))  __bf16 v8bf;
typedef __attribute__((ext_vector_type(2)))  __bf16 v2bf;
typedef __attribute__((ext_vector_type(8)))  float  v8f;
typedef __attribute__((ext_vector_type(4)))  int    v4i;
typedef __attribute__((ext_vector_type(8)))  int    v8i;
typedef __attribute__((ext_vector_type(4)))  unsigned int v4u;

// ------------------------------------------------------------ async copy
#if defined(__has_builtin)
#if __has_builtin(__builtin_amdgcn_global_load_async_to_lds_b128)
#define HAVE_ASYNC 1
#endif
#if __has_builtin(__builtin_amdgcn_tensor_load_to_lds)
#define HAVE_TDM 1
#endif
#endif
#ifndef HAVE_ASYNC
#define HAVE_ASYNC 0
#endif
#ifndef HAVE_TDM
#define HAVE_TDM 0
#endif

#if HAVE_ASYNC
__device__ __forceinline__ void async_b128(const void* g, void* l) {
  __builtin_amdgcn_global_load_async_to_lds_b128((v4i*)g, (v4i*)l, 0, 0);
}
__device__ __forceinline__ void wait_async() {
#if __has_builtin(__builtin_amdgcn_s_wait_asynccnt)
  __builtin_amdgcn_s_wait_asynccnt(0);
#else
  asm volatile("s_wait_asynccnt 0x0" ::: "memory");
#endif
}
#endif

#if HAVE_TDM
__device__ __forceinline__ unsigned lds_addr_of(const void* p) {
  return (unsigned)(size_t)(__attribute__((address_space(3))) const char*)p;
}
// 2-D tile DMA: global [tdim1 rows x tdim0 elems, row stride stride0] tile
// (tile1 x tile0) -> LDS at lds_off, bf16 elems, LDS rows padded per
// pad_interval/pad_amount codes (ISA 8.4).
__device__ __forceinline__ void tdm_load_2d(
    const void* gptr, unsigned lds_off,
    unsigned tdim0, unsigned tdim1, unsigned stride0,
    unsigned tile0, unsigned tile1,
    unsigned pad_interval, unsigned pad_amount) {
  unsigned long long ga = (unsigned long long)(size_t)gptr;
  v4u g0;
  g0[0] = 1u;                                   // count=1, user descriptor
  g0[1] = lds_off;                              // lds_addr
  g0[2] = (unsigned)ga;                         // global_addr[31:0]
  g0[3] = (unsigned)(ga >> 32) | 0x80000000u;   // global_addr[56:32] | type=2
  v8i g1;
  g1[0] = (int)((1u << 16)                      // data_size = 2 bytes
                | (1u << 20)                    // pad_enable
                | (pad_interval << 22) | (pad_amount << 25));
  g1[1] = (int)((tdim0 & 0xFFFFu) << 16);                       // tensor_dim0 lo
  g1[2] = (int)((tdim0 >> 16) | ((tdim1 & 0xFFFFu) << 16));     // dim0 hi|dim1 lo
  g1[3] = (int)((tdim1 >> 16) | (tile0 << 16));                 // dim1 hi|tile0
  g1[4] = (int)tile1;                                           // tile1 (tile2=0)
  g1[5] = (int)stride0;                                         // dim0_stride lo
  g1[6] = 0;
  g1[7] = 0;
  v4i z4 = {0, 0, 0, 0};
  v8i z8 = {0, 0, 0, 0, 0, 0, 0, 0};
  // 6-arg form (amdgpu-toolchain / clang-23): groups 0..3 + extra + cpol
  __builtin_amdgcn_tensor_load_to_lds(g0, g1, z4, z4, z8, 0);
}
__device__ __forceinline__ void wait_tensor() {
  __builtin_amdgcn_s_wait_tensorcnt(0);
}
#endif

// ---------------------------------------------------------------- helpers

__device__ __forceinline__ bf16 f2bf(float f) { return (bf16)f; }

__device__ __forceinline__ unsigned pack2(float lo, float hi) {
  v2bf t = {f2bf(lo), f2bf(hi)};                // -> single v_cvt_pk_bf16_f32
  return __builtin_bit_cast(unsigned, t);
}

__device__ __forceinline__ v16bf combine(v8bf lo, v8bf hi) {
  return __builtin_shufflevector(lo, hi, 0, 1, 2, 3, 4, 5, 6, 7,
                                 8, 9, 10, 11, 12, 13, 14, 15);
}

__device__ __forceinline__ v8f wmma_bf16(v16bf a, v16bf b, v8f c) {
  return __builtin_amdgcn_wmma_f32_16x16x32_bf16(
      false, a, false, b, (short)0, c, false, false);
}

// A-fragment (16x32 bf16) from row-major [m][k] LDS tile.
__device__ __forceinline__ v16bf frag_a(const bf16* s, int ld, int row0, int lane) {
  int m  = row0 + (lane & 15);
  int kb = (lane & 16) >> 1;                 // 0 or 8
  const bf16* p = s + m * ld + kb;
  return combine(*(const v8bf*)(p), *(const v8bf*)(p + 16));
}

// B-fragment (32x16 bf16) from *transposed* [n][k] LDS tile.
__device__ __forceinline__ v16bf frag_bt(const bf16* s, int ld, int col0, int lane) {
  int n   = col0 + (lane & 15);
  int klo = (lane & 16) >> 1;                // 0 or 8
  const bf16* p = s + n * ld + klo;
  return combine(*(const v8bf*)(p), *(const v8bf*)(p + 16));
}

// ---------------------------------------------------------------- QKV GEMM

#define LDA 40      // 32 + 8 pad (bf16), 80 B row stride
#define LDK 72      // 64 + 8 pad (bf16), 144 B row stride

__global__ __launch_bounds__(256) void qkv_gemm(
    const float* __restrict__ X, const float* __restrict__ W,
    const float* __restrict__ bias,
    bf16* __restrict__ Q, bf16* __restrict__ K, bf16* __restrict__ V) {
  __shared__ bf16 As[64 * LDA];
  __shared__ bf16 Bt[256 * LDA];             // transposed: [n][k]
  const int tid = threadIdx.x, lane = tid & 31, wid = tid >> 5;
  const int wm = wid & 1, wn = wid >> 1;
  const int m0 = blockIdx.x * 64;
  const int n0 = blockIdx.y * 256;

  v8f acc[2][4];
  const v8f vzero = {0.f, 0.f, 0.f, 0.f, 0.f, 0.f, 0.f, 0.f};
#pragma unroll
  for (int i = 0; i < 2; ++i)
#pragma unroll
    for (int j = 0; j < 4; ++j) acc[i][j] = vzero;

  for (int k0 = 0; k0 < 1024; k0 += 32) {
    if (k0 + 32 < 1024) __builtin_prefetch(&X[(m0 + (tid >> 2)) * 1024 + k0 + 32]);
    // stage A: 64x32 fp32 -> bf16, float4 chunks
#pragma unroll
    for (int ci = tid; ci < 512; ci += 256) {
      int r = ci >> 3, c4 = (ci & 7) << 2;
      const float4 xv = *(const float4*)&X[(m0 + r) * 1024 + k0 + c4];
      uint2 pk;
      pk.x = pack2(xv.x, xv.y);
      pk.y = pack2(xv.z, xv.w);
      *(uint2*)&As[r * LDA + c4] = pk;
    }
    // stage B transposed: 32x256 fp32 -> Bt[n][k]
#pragma unroll
    for (int ci = tid; ci < 2048; ci += 256) {
      int n = ci & 255, kb = (ci >> 8) << 2;
      float w0 = W[(k0 + kb + 0) * 3072 + n0 + n];
      float w1 = W[(k0 + kb + 1) * 3072 + n0 + n];
      float w2 = W[(k0 + kb + 2) * 3072 + n0 + n];
      float w3 = W[(k0 + kb + 3) * 3072 + n0 + n];
      uint2 pk;
      pk.x = pack2(w0, w1);
      pk.y = pack2(w2, w3);
      *(uint2*)&Bt[n * LDA + kb] = pk;
    }
    __syncthreads();

    v16bf af[2], bfr[4];
#pragma unroll
    for (int mi = 0; mi < 2; ++mi) af[mi] = frag_a(As, LDA, wm * 32 + mi * 16, lane);
#pragma unroll
    for (int ni = 0; ni < 4; ++ni) bfr[ni] = frag_bt(Bt, LDA, wn * 64 + ni * 16, lane);
#pragma unroll
    for (int mi = 0; mi < 2; ++mi)
#pragma unroll
      for (int ni = 0; ni < 4; ++ni)
        acc[mi][ni] = wmma_bf16(af[mi], bfr[ni], acc[mi][ni]);
    __syncthreads();
  }

#pragma unroll
  for (int mi = 0; mi < 2; ++mi)
#pragma unroll
    for (int ni = 0; ni < 4; ++ni) {
      int col   = n0 + wn * 64 + ni * 16 + (lane & 15);
      float bv  = bias[col];
      int which = col >> 10;                 // 0=q 1=k 2=v
      int cc    = col & 1023;
      int h     = cc >> 6, d = cc & 63;
      float sc  = (which == 0) ? 0.125f : 1.0f;   // fold 1/sqrt(64) into Q
      bf16* dst = (which == 0) ? Q : ((which == 1) ? K : V);
#pragma unroll
      for (int r = 0; r < 8; ++r) {
        int row = m0 + wm * 32 + mi * 16 + r + ((lane & 16) >> 1);
        int bb  = row >> 11, t = row & 2047;
        float v = (acc[mi][ni][r] + bv) * sc;
        size_t idx = (which == 2)
            ? ((size_t)((bb * 16 + h) * 64 + d) << 11) + t       // V: [B,H,D,T]
            : ((size_t)((bb * 16 + h) * 2048 + t) << 6) + d;     // Q,K: [B,H,T,D]
        dst[idx] = f2bf(v);
      }
    }
}

// ------------------------------------------------------------- attention

__global__ __launch_bounds__(256) void attn_kernel(
    const bf16* __restrict__ Q, const bf16* __restrict__ K,
    const bf16* __restrict__ V, bf16* __restrict__ Y) {
  __shared__ bf16 Ks[64 * LDK];              // [k][d]
  __shared__ bf16 Vt[64 * LDK];              // [d][k]  (V is [B,H,D,T] in global)
  __shared__ bf16 Ps[8 * 16 * LDK];
  const int tid = threadIdx.x, lane = tid & 31, w = tid >> 5;
  const int qt = blockIdx.x, h = blockIdx.y, b = blockIdx.z;
  const int q0 = qt * 128;
  const int qw = q0 + w * 16;
  const bf16* Qb = Q + ((size_t)(b * 16 + h) << 11) * 64;
  const bf16* Kb = K + ((size_t)(b * 16 + h) << 11) * 64;
  const bf16* Vb = V + ((size_t)(b * 16 + h) << 6) * 2048;
  bf16* Pw = Ps + w * 16 * LDK;

  v16bf qf[2];
  {
    int m  = qw + (lane & 15);
    int kb = (lane & 16) >> 1;
    const bf16* p = Qb + m * 64 + kb;
    qf[0] = combine(*(const v8bf*)(p), *(const v8bf*)(p + 16));
    qf[1] = combine(*(const v8bf*)(p + 32), *(const v8bf*)(p + 48));
  }

  v8f o[4];
  const v8f vzero = {0.f, 0.f, 0.f, 0.f, 0.f, 0.f, 0.f, 0.f};
#pragma unroll
  for (int c = 0; c < 4; ++c) o[c] = vzero;
  float mrun[8], lrun[8];
#pragma unroll
  for (int r = 0; r < 8; ++r) { mrun[r] = -3.0e38f; lrun[r] = 0.0f; }

  const int ktiles = (q0 + 128) >> 6;
  for (int kt = 0; kt < ktiles; ++kt) {
    int k0 = kt * 64;
    const bf16* ksrc = Kb + k0 * 64;          // K rows contiguous in [T,D]
    const bf16* vsrc = Vb + k0;               // V rows (d) stride 2048 in [D,T]
#if HAVE_TDM
    if (w == 0) {                             // one TDM issue per block (wave 0)
      // K tile: 64x64 from [2048,64]; pad 128B rows by 16B (codes 4,3)
      tdm_load_2d(ksrc, lds_addr_of(Ks), 64, 2048, 64, 64, 64, 4, 3);
      // V tile: 64x64 from [64,2048] (d-major); same padding
      tdm_load_2d(vsrc, lds_addr_of(Vt), 2048, 64, 2048, 64, 64, 4, 3);
      wait_tensor();
    }
#elif HAVE_ASYNC
#pragma unroll
    for (int ci = tid; ci < 512; ci += 256) { // 16 B chunks
      int r = ci >> 3, c8 = (ci & 7) << 3;
      async_b128(ksrc + r * 64 + c8,   &Ks[r * LDK + c8]);
      async_b128(vsrc + r * 2048 + c8, &Vt[r * LDK + c8]);
    }
    wait_async();
#else
#pragma unroll
    for (int ci = tid; ci < 512; ci += 256) {
      int r = ci >> 3, c8 = (ci & 7) << 3;
      *(v8bf*)&Ks[r * LDK + c8] = *(const v8bf*)(ksrc + r * 64 + c8);
      *(v8bf*)&Vt[r * LDK + c8] = *(const v8bf*)(vsrc + r * 2048 + c8);
    }
#endif
    __syncthreads();

    if (k0 <= qw + 15) {                      // wave-uniform -> EXEC stays full
      v8f s[4];
#pragma unroll
      for (int c = 0; c < 4; ++c) s[c] = vzero;
#pragma unroll
      for (int c = 0; c < 4; ++c)
#pragma unroll
        for (int half = 0; half < 2; ++half)
          s[c] = wmma_bf16(qf[half], frag_bt(Ks + half * 32, LDK, c * 16, lane), s[c]);

      // causal mask
#pragma unroll
      for (int c = 0; c < 4; ++c) {
        int col = k0 + c * 16 + (lane & 15);
#pragma unroll
        for (int r = 0; r < 8; ++r) {
          int row = qw + r + ((lane & 16) >> 1);
          if (col > row) s[c][r] = -3.0e38f;
        }
      }

      // online softmax (rows live in 16-lane half-wave groups)
      float alpha[8];
#pragma unroll
      for (int r = 0; r < 8; ++r) {
        float mrow = fmaxf(fmaxf(s[0][r], s[1][r]), fmaxf(s[2][r], s[3][r]));
#pragma unroll
        for (int msk = 1; msk < 16; msk <<= 1)
          mrow = fmaxf(mrow, __shfl_xor(mrow, msk, 32));
        float mnew = fmaxf(mrun[r], mrow);
        alpha[r] = __expf(mrun[r] - mnew);
        mrun[r]  = mnew;
        float psum = 0.0f;
#pragma unroll
        for (int c = 0; c < 4; ++c) {
          float p = __expf(s[c][r] - mnew);
          s[c][r] = p;
          psum += p;
        }
#pragma unroll
        for (int msk = 1; msk < 16; msk <<= 1)
          psum += __shfl_xor(psum, msk, 32);
        lrun[r] = lrun[r] * alpha[r] + psum;
#pragma unroll
        for (int c = 0; c < 4; ++c) o[c][r] *= alpha[r];
      }

      // C-layout P -> A-layout via per-wave LDS bounce
#pragma unroll
      for (int c = 0; c < 4; ++c)
#pragma unroll
        for (int r = 0; r < 8; ++r)
          Pw[(r + ((lane & 16) >> 1)) * LDK + c * 16 + (lane & 15)] = f2bf(s[c][r]);

      // O += P @ V
#pragma unroll
      for (int half = 0; half < 2; ++half) {
        v16bf pa = frag_a(Pw + half * 32, LDK, 0, lane);
#pragma unroll
        for (int c2 = 0; c2 < 4; ++c2)
          o[c2] = wmma_bf16(pa, frag_bt(Vt + half * 32, LDK, c2 * 16, lane), o[c2]);
      }
    }
    __syncthreads();
  }

#pragma unroll
  for (int c2 = 0; c2 < 4; ++c2) {
    int d = c2 * 16 + (lane & 15);
#pragma unroll
    for (int r = 0; r < 8; ++r) {
      int row = qw + r + ((lane & 16) >> 1);
      float v = o[c2][r] / lrun[r];
      Y[((size_t)(b * 2048 + row) << 10) + h * 64 + d] = f2bf(v);
    }
  }
}

// --------------------------------------------------------------- proj GEMM

__global__ __launch_bounds__(256) void proj_gemm(
    const bf16* __restrict__ Yb, const float* __restrict__ W,
    const float* __restrict__ bias, float* __restrict__ out) {
  __shared__ bf16 As[64 * LDA];
  __shared__ bf16 Bt[256 * LDA];             // transposed [n][k]
  const int tid = threadIdx.x, lane = tid & 31, wid = tid >> 5;
  const int wm = wid & 1, wn = wid >> 1;
  const int m0 = blockIdx.x * 64;
  const int n0 = blockIdx.y * 256;

  v8f acc[2][4];
  const v8f vzero = {0.f, 0.f, 0.f, 0.f, 0.f, 0.f, 0.f, 0.f};
#pragma unroll
  for (int i = 0; i < 2; ++i)
#pragma unroll
    for (int j = 0; j < 4; ++j) acc[i][j] = vzero;

  for (int k0 = 0; k0 < 1024; k0 += 32) {
    if (k0 + 32 < 1024) __builtin_prefetch(&Yb[(m0 + (tid >> 2)) * 1024 + k0 + 32]);
    // stage A (bf16 64x32 tile)
#if HAVE_TDM
    if (wid == 0)                               // 64B rows + 16B pad: codes 3,3
      tdm_load_2d(&Yb[(size_t)m0 * 1024 + k0], lds_addr_of(As),
                  1024, 8192, 1024, 32, 64, 3, 3);
#elif HAVE_ASYNC
#pragma unroll
    for (int ci = tid; ci < 256; ci += 256) {
      int r = ci >> 2, c8 = (ci & 3) << 3;
      async_b128(&Yb[(m0 + r) * 1024 + k0 + c8], &As[r * LDA + c8]);
    }
#else
#pragma unroll
    for (int ci = tid; ci < 256; ci += 256) {
      int r = ci >> 2, c8 = (ci & 3) << 3;
      *(v8bf*)&As[r * LDA + c8] = *(const v8bf*)&Yb[(m0 + r) * 1024 + k0 + c8];
    }
#endif
    // stage B transposed: fp32 -> Bt[n][k]
#pragma unroll
    for (int ci = tid; ci < 2048; ci += 256) {
      int n = ci & 255, kb = (ci >> 8) << 2;
      float w0 = W[(k0 + kb + 0) * 1024 + n0 + n];
      float w1 = W[(k0 + kb + 1) * 1024 + n0 + n];
      float w2 = W[(k0 + kb + 2) * 1024 + n0 + n];
      float w3 = W[(k0 + kb + 3) * 1024 + n0 + n];
      uint2 pk;
      pk.x = pack2(w0, w1);
      pk.y = pack2(w2, w3);
      *(uint2*)&Bt[n * LDA + kb] = pk;
    }
#if HAVE_TDM
    if (wid == 0) wait_tensor();
#elif HAVE_ASYNC
    wait_async();
#endif
    __syncthreads();

    v16bf af[2], bfr[4];
#pragma unroll
    for (int mi = 0; mi < 2; ++mi) af[mi] = frag_a(As, LDA, wm * 32 + mi * 16, lane);
#pragma unroll
    for (int ni = 0; ni < 4; ++ni) bfr[ni] = frag_bt(Bt, LDA, wn * 64 + ni * 16, lane);
#pragma unroll
    for (int mi = 0; mi < 2; ++mi)
#pragma unroll
      for (int ni = 0; ni < 4; ++ni)
        acc[mi][ni] = wmma_bf16(af[mi], bfr[ni], acc[mi][ni]);
    __syncthreads();
  }

#pragma unroll
  for (int mi = 0; mi < 2; ++mi)
#pragma unroll
    for (int ni = 0; ni < 4; ++ni) {
      int col  = n0 + wn * 64 + ni * 16 + (lane & 15);
      float bv = bias[col];
#pragma unroll
      for (int r = 0; r < 8; ++r) {
        int row = m0 + wm * 32 + mi * 16 + r + ((lane & 16) >> 1);
        out[(size_t)row * 1024 + col] = acc[mi][ni][r] + bv;
      }
    }
}

// ---------------------------------------------------------------- launch

extern "C" void kernel_launch(void* const* d_in, const int* in_sizes, int n_in,
                              void* d_out, int out_size, void* d_ws, size_t ws_size,
                              hipStream_t stream) {
  const float* x      = (const float*)d_in[0];
  const float* W_attn = (const float*)d_in[1];
  const float* b_attn = (const float*)d_in[2];
  const float* W_proj = (const float*)d_in[3];
  const float* b_proj = (const float*)d_in[4];
  float* out = (float*)d_out;

  const size_t QKV_ELEMS = (size_t)4 * 16 * 2048 * 64;   // 8,388,608
  bf16* Qw = (bf16*)d_ws;
  bf16* Kw = Qw + QKV_ELEMS;
  bf16* Vw = Kw + QKV_ELEMS;
  bf16* Yw = Vw + QKV_ELEMS;

  qkv_gemm<<<dim3(128, 12), 256, 0, stream>>>(x, W_attn, b_attn, Qw, Kw, Vw);
  attn_kernel<<<dim3(16, 16, 4), 256, 0, stream>>>(Qw, Kw, Vw, Yw);
  proj_gemm<<<dim3(128, 4), 256, 0, stream>>>(Yw, W_proj, b_proj, out);
}
